// moe_attention_27101243638232
// MI455X (gfx1250) — matrix operand order repaired and checked by reading, not compiled, not run on hardware
//
#include <hip/hip_runtime.h>
#include <hip/hip_bf16.h>

// ---------------------------------------------------------------------------
// Types for CDNA5 WMMA (wave32, 16x16x32 bf16 -> f32)
// ---------------------------------------------------------------------------
typedef __attribute__((ext_vector_type(16))) __bf16 v16bf;
typedef __attribute__((ext_vector_type(8)))  float  v8f;
typedef __attribute__((ext_vector_type(4)))  unsigned int u32x4;
typedef __attribute__((ext_vector_type(4)))  unsigned int tdm_g0_t;
typedef __attribute__((ext_vector_type(8)))  int          tdm_g1_t;
typedef __attribute__((ext_vector_type(4)))  int          tdm_g2_t;
typedef __attribute__((ext_vector_type(8)))  int          tdm_g4_t;

union FragAB {
    v16bf v;
    u32x4 q[2];
};

__device__ __forceinline__ unsigned short f2bf(float f) {
    union { float f; unsigned u; } x;
    x.f = f;
    unsigned u = x.u;
    u += 0x7fffu + ((u >> 16) & 1u);   // round-to-nearest-even
    return (unsigned short)(u >> 16);
}

__device__ __forceinline__ v8f wmma_bf16(const FragAB& a, const FragAB& b, v8f c) {
    return __builtin_amdgcn_wmma_f32_16x16x32_bf16(
        false, a.v, false, b.v, (short)0, c, false, false);
}

// A-matrix fragment (16x32 bf16): lane holds row m=lane%16; g=lane/16.
// VGPR halves: K in [k0+8g, k0+8g+7] and [k0+16+8g, k0+16+8g+7] (contiguous).
__device__ __forceinline__ FragAB load_a_frag(const unsigned short* row_base,
                                              int k0, int lane) {
    FragAB f;
    int g = lane >> 4;
    f.q[0] = *(const u32x4*)(row_base + k0 + 8 * g);
    f.q[1] = *(const u32x4*)(row_base + k0 + 16 + 8 * g);
    return f;
}

// B-matrix fragment (32x16 bf16) from a [N][K]-contiguous ("B-transposed") tile:
// lane holds col n=lane%16; K in [k0+8g, k0+8g+7] and [k0+16+8g, k0+16+8g+7].
__device__ __forceinline__ FragAB load_bt_frag(const unsigned short* t,
                                               int row_stride, int n_base,
                                               int k0, int lane) {
    FragAB f;
    int n = lane & 15, g = lane >> 4;
    const unsigned short* p = t + (n_base + n) * row_stride + k0 + 8 * g;
    f.q[0] = *(const u32x4*)(p);
    f.q[1] = *(const u32x4*)(p + 16);
    return f;
}

// ---------------------------------------------------------------------------
// Tensor Data Mover descriptor helpers (CDNA5 D#, CDNA5 ISA)
// data_size = 4 bytes (code 2); 2D tile with LDS padding.
// ---------------------------------------------------------------------------
#if __has_builtin(__builtin_amdgcn_tensor_load_to_lds) && \
    __has_builtin(__builtin_amdgcn_s_wait_tensorcnt)
#define USE_TDM 1
#else
#define USE_TDM 0
#endif

__device__ __forceinline__ tdm_g0_t tdm_make_g0(unsigned lds_addr,
                                                unsigned long long gaddr) {
    tdm_g0_t g;
    g[0] = 1u;                                       // count=1, user mode
    g[1] = lds_addr;                                 // bits 63:32
    g[2] = (unsigned)(gaddr & 0xFFFFFFFFull);        // global_addr[31:0]
    g[3] = (unsigned)((gaddr >> 32) & 0x1FFFFFFull)  // global_addr[56:32]
           | (2u << 30);                             // type=2 ("image")
    return g;
}

// 2D tile: tile0 x tile1 (data_size units), tensor dim0/dim1 sized to the tile,
// dim0 stride in data_size units. Pad: interval code c -> 2<<c DWORDs,
// amount code a -> (a+1) DWORDs.
__device__ __forceinline__ tdm_g1_t tdm_make_g1(unsigned tdim0, unsigned tdim1,
                                                unsigned tile0, unsigned tile1,
                                                unsigned long long stride0,
                                                unsigned pad_int_code,
                                                unsigned pad_amt_code) {
    tdm_g1_t g;
    g[0] = (int)((2u << 16)                 // data_size = 4 bytes
                 | (1u << 20)               // pad_enable
                 | (pad_int_code << 22)
                 | (pad_amt_code << 25));
    g[1] = (int)((tdim0 & 0xFFFFu) << 16);
    g[2] = (int)((tdim0 >> 16) | ((tdim1 & 0xFFFFu) << 16));
    g[3] = (int)((tdim1 >> 16) | (tile0 << 16));
    g[4] = (int)(tile1 & 0xFFFFu);                    // tile_dim1; tile_dim2=0
    g[5] = (int)(stride0 & 0xFFFFFFFFull);
    g[6] = (int)((stride0 >> 32) & 0xFFFFull);        // dim1_stride unused (2D)
    g[7] = 0;
    return g;
}

#if USE_TDM
#define TDM_LOAD(lds, gaddr, g1)                                              \
    __builtin_amdgcn_tensor_load_to_lds(tdm_make_g0((lds), (gaddr)), (g1),    \
                                        (tdm_g2_t){0, 0, 0, 0},               \
                                        (tdm_g2_t){0, 0, 0, 0},               \
                                        (tdm_g4_t){0, 0, 0, 0, 0, 0, 0, 0}, 0)
#endif

// ---------------------------------------------------------------------------
// fp32 -> bf16 conversion (RNE), 4 elements / thread
// ---------------------------------------------------------------------------
__global__ void cvt_f32_bf16(const float* __restrict__ in,
                             unsigned short* __restrict__ out, int n) {
    int i = (blockIdx.x * 256 + threadIdx.x) * 4;
    if (i >= n) return;
    float4 f = *(const float4*)(in + i);
    unsigned short r0 = f2bf(f.x), r1 = f2bf(f.y), r2 = f2bf(f.z), r3 = f2bf(f.w);
    unsigned lo = (unsigned)r0 | ((unsigned)r1 << 16);
    unsigned hi = (unsigned)r2 | ((unsigned)r3 << 16);
    uint2 pk; pk.x = lo; pk.y = hi;
    *(uint2*)(out + i) = pk;
}

// ---------------------------------------------------------------------------
// fp32 [K][N] -> bf16 [N][K] transpose+convert, 64x64 LDS tiles
// ---------------------------------------------------------------------------
__global__ __launch_bounds__(256, 1)
void cvtT_f32_bf16(const float* __restrict__ in, unsigned short* __restrict__ out,
                   int K, int N) {
    __shared__ unsigned short T[64 * 72];
    const int tid = threadIdx.x;
    const int k0 = blockIdx.y * 64, n0 = blockIdx.x * 64;
#pragma unroll
    for (int it = 0; it < 4; ++it) {
        int idx = tid + it * 256;                 // 0..1023
        int r = idx >> 4, c = (idx & 15) * 4;     // 64 rows x 16 float4
        float4 f = *(const float4*)(in + (size_t)(k0 + r) * N + n0 + c);
        T[r * 72 + c + 0] = f2bf(f.x);
        T[r * 72 + c + 1] = f2bf(f.y);
        T[r * 72 + c + 2] = f2bf(f.z);
        T[r * 72 + c + 3] = f2bf(f.w);
    }
    __syncthreads();
#pragma unroll
    for (int it = 0; it < 2; ++it) {
        int idx = tid + it * 256;                 // 0..511
        int n = idx >> 3, c = (idx & 7) * 8;      // 64 n-rows x 8 chunks of 8
        union { unsigned short s[8]; u32x4 v; } tmp;
#pragma unroll
        for (int j = 0; j < 8; ++j) tmp.s[j] = T[(c + j) * 72 + n];
        *(u32x4*)(out + (size_t)(n0 + n) * K + k0 + c) = tmp.v;
    }
}

// ---------------------------------------------------------------------------
// Tiled bf16 GEMM: OUT[M,N] = A[M,K] @ Bt[N,K]^T   (Bt is pre-transposed)
// 128x128 macro tile, 8 waves (4 in M x 2 in N), each wave 2x4 WMMA frags.
// Double-buffered TDM pipeline: wave0 issues next tile pair, waits
// tensorcnt<=2 (current pair landed, next pair in flight), barrier, compute.
// OUT_MODE 0: bf16 output.  OUT_MODE 1: fp32 output + bias.
// ---------------------------------------------------------------------------
template <int OUT_MODE>
__global__ __launch_bounds__(256, 1)
void gemm_bf16(const unsigned short* __restrict__ A,
               const unsigned short* __restrict__ Bt,
               void* __restrict__ OutPtr,
               const float* __restrict__ bias,
               int M, int N, int K) {
    __shared__ unsigned short As[2][128 * 40];   // [m][k], pad 32->40
    __shared__ unsigned short Bs[2][128 * 40];   // [n][k], pad 32->40

    const int tid = threadIdx.x;
    const int lane = tid & 31, wid = tid >> 5;
    const int wm = wid & 3, wn = wid >> 2;           // 4 x 2 wave grid
    const int m0 = blockIdx.y * 128, n0 = blockIdx.x * 128;

    v8f acc[2][4];
#pragma unroll
    for (int i = 0; i < 2; ++i)
#pragma unroll
        for (int j = 0; j < 4; ++j)
#pragma unroll
            for (int e = 0; e < 8; ++e) acc[i][j][e] = 0.0f;

#if USE_TDM
    // 2D tile: 16 dwords wide x 128 rows; pad interval 16 DW (code 3),
    // pad amount 4 DW (code 3) -> LDS pitch 20 DW = 40 ushorts.
    const tdm_g1_t g1 = tdm_make_g1(16u, 128u, 16u, 128u,
                                    (unsigned long long)(K >> 1), 3u, 3u);
    const unsigned ldsA0 = (unsigned)(unsigned long long)(void*)&As[0][0];
    const unsigned ldsA1 = (unsigned)(unsigned long long)(void*)&As[1][0];
    const unsigned ldsB0 = (unsigned)(unsigned long long)(void*)&Bs[0][0];
    const unsigned ldsB1 = (unsigned)(unsigned long long)(void*)&Bs[1][0];

    if (wid == 0) {   // prologue: fill buffer 0
        TDM_LOAD(ldsA0, (unsigned long long)(A + (size_t)m0 * K), g1);
        TDM_LOAD(ldsB0, (unsigned long long)(Bt + (size_t)n0 * K), g1);
    }
    int cur = 0;
    for (int kt = 0; kt < K; kt += 32, cur ^= 1) {
        if (wid == 0) {
            if (kt + 32 < K) {   // prefetch next pair into the other buffer
                TDM_LOAD(cur ? ldsA0 : ldsA1,
                         (unsigned long long)(A + (size_t)m0 * K + kt + 32), g1);
                TDM_LOAD(cur ? ldsB0 : ldsB1,
                         (unsigned long long)(Bt + (size_t)n0 * K + kt + 32), g1);
                __builtin_amdgcn_s_wait_tensorcnt(2);  // current pair landed
            } else {
                __builtin_amdgcn_s_wait_tensorcnt(0);
            }
        }
        __syncthreads();

        FragAB af[2];
#pragma unroll
        for (int im = 0; im < 2; ++im)
            af[im] = load_a_frag(&As[cur][(wm * 32 + im * 16 + (lane & 15)) * 40],
                                 0, lane);
        FragAB bfr[4];
#pragma unroll
        for (int f = 0; f < 4; ++f)
            bfr[f] = load_bt_frag(&Bs[cur][0], 40, wn * 64 + f * 16, 0, lane);

#pragma unroll
        for (int im = 0; im < 2; ++im)
#pragma unroll
            for (int f = 0; f < 4; ++f)
                acc[im][f] = wmma_bf16(af[im], bfr[f], acc[im][f]);
        __syncthreads();   // all reads of buffer `cur` done before its reuse
    }
#else
    for (int kt = 0; kt < K; kt += 32) {
        // Fallback: coalesced 16B loads (both A and Bt are k-contiguous)
#pragma unroll
        for (int it = 0; it < 2; ++it) {
            int idx = tid + it * 256;                // 0..511
            int r = idx >> 2, c = (idx & 3) * 8;
            *(u32x4*)(&As[0][r * 40 + c]) =
                *(const u32x4*)(A + (size_t)(m0 + r) * K + kt + c);
            *(u32x4*)(&Bs[0][r * 40 + c]) =
                *(const u32x4*)(Bt + (size_t)(n0 + r) * K + kt + c);
        }
        __syncthreads();
        FragAB af[2];
#pragma unroll
        for (int im = 0; im < 2; ++im)
            af[im] = load_a_frag(&As[0][(wm * 32 + im * 16 + (lane & 15)) * 40],
                                 0, lane);
        FragAB bfr[4];
#pragma unroll
        for (int f = 0; f < 4; ++f)
            bfr[f] = load_bt_frag(&Bs[0][0], 40, wn * 64 + f * 16, 0, lane);
#pragma unroll
        for (int im = 0; im < 2; ++im)
#pragma unroll
            for (int f = 0; f < 4; ++f)
                acc[im][f] = wmma_bf16(af[im], bfr[f], acc[im][f]);
        __syncthreads();
    }
#endif

    // Epilogue. C/D layout: lane holds col n=lane%16; VGPR r holds row r + 8*(lane/16)
    const int g = lane >> 4, nc = lane & 15;
#pragma unroll
    for (int im = 0; im < 2; ++im)
#pragma unroll
        for (int f = 0; f < 4; ++f)
#pragma unroll
            for (int r = 0; r < 8; ++r) {
                int row = m0 + wm * 32 + im * 16 + g * 8 + r;
                int col = n0 + wn * 64 + f * 16 + nc;
                if (OUT_MODE == 0) {
                    ((unsigned short*)OutPtr)[(size_t)row * N + col] = f2bf(acc[im][f][r]);
                } else {
                    ((float*)OutPtr)[(size_t)row * N + col] = acc[im][f][r] + bias[col];
                }
            }
}

// ---------------------------------------------------------------------------
// Flash attention (non-causal), Dh=64, online softmax.
// Block: 128 query rows, 8 waves x 16 rows. Key blocks of 64.
// Q and K tiles DMA'd by the Tensor Data Mover (pad -> 72-ushort pitch);
// V is loaded+transposed manually by all threads while the K DMA runs.
// qvbuf: [B*N, 2048] bf16 (q at col h*64, v at col 1024 + h*64)
// kbuf : [B*N, 1024] bf16 (k at col h*64)
// xbuf : [B*N, 1024] bf16 output (attention result, head-merged)
// ---------------------------------------------------------------------------
__global__ __launch_bounds__(256, 1)
void flash_attn(const unsigned short* __restrict__ qvbuf,
                const unsigned short* __restrict__ kbuf,
                unsigned short* __restrict__ xbuf,
                int Nctx) {
    __shared__ unsigned short Qs[128 * 72];      // [qrow][d], pad 64->72
    __shared__ unsigned short Ks[64 * 72];       // [krow][d]   (== B^T for QK^T)
    __shared__ unsigned short Vt[64 * 72];       // [d][krow]   (== B^T for P@V)
    __shared__ unsigned short Ps[8 * 16 * 72];   // per-wave P staging [16][72]

    const int tid = threadIdx.x, lane = tid & 31, wid = tid >> 5;
    const int b = blockIdx.y >> 4, h = blockIdx.y & 15;
    const int q0 = blockIdx.x * 128;
    const int qcol = h * 64, vcol = 1024 + h * 64;
    const size_t qrow0 = (size_t)(b * Nctx + q0);
    const float scale = 0.125f;                   // Dh^-0.5, Dh=64

#if USE_TDM
    // tile rows of 32 DW, pad interval 32 DW (code 4), amount 4 DW (code 3)
    const tdm_g1_t g1q = tdm_make_g1(32u, 128u, 32u, 128u, 1024ull, 4u, 3u);
    const tdm_g1_t g1k = tdm_make_g1(32u, 64u, 32u, 64u, 512ull, 4u, 3u);
    const unsigned ldsQ = (unsigned)(unsigned long long)(void*)Qs;
    const unsigned ldsK = (unsigned)(unsigned long long)(void*)Ks;
    if (wid == 0) {
        TDM_LOAD(ldsQ, (unsigned long long)(qvbuf + qrow0 * 2048 + qcol), g1q);
        __builtin_amdgcn_s_wait_tensorcnt(0);
    }
#else
    // Load Q block 128x64 manually
#pragma unroll
    for (int it = 0; it < 4; ++it) {
        int idx = tid + it * 256;                 // 0..1023
        int r = idx >> 3, c = (idx & 7) * 8;
        u32x4 d = *(const u32x4*)(qvbuf + (qrow0 + r) * 2048 + qcol + c);
        *(u32x4*)(Qs + r * 72 + c) = d;
    }
#endif
    __syncthreads();

    v8f o[4];
    float mrow[8], lrow[8];
#pragma unroll
    for (int f = 0; f < 4; ++f)
#pragma unroll
        for (int e = 0; e < 8; ++e) o[f][e] = 0.0f;
#pragma unroll
    for (int r = 0; r < 8; ++r) { mrow[r] = -3.0e38f; lrow[r] = 0.0f; }

    const int g = lane >> 4;
    const unsigned short* qrow_base = Qs + (wid * 16 + (lane & 15)) * 72;

    for (int j0 = 0; j0 < Nctx; j0 += 64) {
#if USE_TDM
        if (wid == 0)   // K tile 64x64 via TDM (overlaps the manual V loads)
            TDM_LOAD(ldsK,
                     (unsigned long long)(kbuf + (size_t)(b * Nctx + j0) * 1024 + qcol),
                     g1k);
#else
        // K tile 64x64 row-major, manual
#pragma unroll
        for (int it = 0; it < 2; ++it) {
            int idx = tid + it * 256;
            int r = idx >> 3, c = (idx & 7) * 8;
            u32x4 d = *(const u32x4*)(kbuf + (size_t)(b * Nctx + j0 + r) * 1024 + qcol + c);
            *(u32x4*)(Ks + r * 72 + c) = d;
        }
#endif
        // V tile 64x64 transposed into Vt[d][krow]
#pragma unroll
        for (int it = 0; it < 2; ++it) {
            int idx = tid + it * 256;
            int r = idx >> 3, c = (idx & 7) * 8;  // r=krow, c=dim base
            u32x4 d = *(const u32x4*)(qvbuf + (size_t)(b * Nctx + j0 + r) * 2048 + vcol + c);
            const unsigned short* s = (const unsigned short*)&d;
#pragma unroll
            for (int j = 0; j < 8; ++j) Vt[(c + j) * 72 + r] = s[j];
        }
#if USE_TDM
        if (wid == 0) __builtin_amdgcn_s_wait_tensorcnt(0);
#endif
        __syncthreads();

        // ---- S = scale * Q Kt : 4 frags of 16x16, K-dim 64 = 2 WMMA each ----
        FragAB qa[2];
        qa[0] = load_a_frag(qrow_base, 0, lane);
        qa[1] = load_a_frag(qrow_base, 32, lane);

        v8f sfr[4];
#pragma unroll
        for (int f = 0; f < 4; ++f) {
#pragma unroll
            for (int e = 0; e < 8; ++e) sfr[f][e] = 0.0f;
#pragma unroll
            for (int kc = 0; kc < 2; ++kc) {
                FragAB kb = load_bt_frag(Ks, 72, f * 16, kc * 32, lane);
                sfr[f] = wmma_bf16(qa[kc], kb, sfr[f]);
            }
#pragma unroll
            for (int e = 0; e < 8; ++e) sfr[f][e] *= scale;
        }

        // ---- online softmax across the 64 keys of this block ----
        unsigned short* pw = Ps + wid * (16 * 72);
#pragma unroll
        for (int r = 0; r < 8; ++r) {
            float mx = fmaxf(fmaxf(sfr[0][r], sfr[1][r]), fmaxf(sfr[2][r], sfr[3][r]));
#pragma unroll
            for (int off = 1; off < 16; off <<= 1)
                mx = fmaxf(mx, __shfl_xor(mx, off, 32));
            float mnew = fmaxf(mrow[r], mx);
            float alpha = __expf(mrow[r] - mnew);
            mrow[r] = mnew;
            float rs = 0.0f;
#pragma unroll
            for (int f = 0; f < 4; ++f) {
                float p = __expf(sfr[f][r] - mnew);
                sfr[f][r] = p;
                rs += p;
            }
#pragma unroll
            for (int off = 1; off < 16; off <<= 1)
                rs += __shfl_xor(rs, off, 32);
            lrow[r] = lrow[r] * alpha + rs;
#pragma unroll
            for (int f = 0; f < 4; ++f) o[f][r] *= alpha;
            // stage P (bf16) in this wave's private LDS region, [row][key] layout
#pragma unroll
            for (int f = 0; f < 4; ++f)
                pw[(r + 8 * g) * 72 + f * 16 + (lane & 15)] = f2bf(sfr[f][r]);
        }

        // ---- O += P V ----
        FragAB pa[2];
        {
            const unsigned short* pb = pw + (lane & 15) * 72;
            pa[0] = load_a_frag(pb, 0, lane);
            pa[1] = load_a_frag(pb, 32, lane);
        }
#pragma unroll
        for (int df = 0; df < 4; ++df) {
#pragma unroll
            for (int kc = 0; kc < 2; ++kc) {
                FragAB vb = load_bt_frag(Vt, 72, df * 16, kc * 32, lane);
                o[df] = wmma_bf16(pa[kc], vb, o[df]);
            }
        }
        __syncthreads();   // before next K/V tile overwrite
    }

    // ---- epilogue: O /= l, write bf16 to xbuf ----
#pragma unroll
    for (int r = 0; r < 8; ++r) {
        float inv = 1.0f / lrow[r];
        int row = b * Nctx + q0 + wid * 16 + r + 8 * g;
#pragma unroll
        for (int df = 0; df < 4; ++df) {
            int col = h * 64 + df * 16 + (lane & 15);
            xbuf[(size_t)row * 1024 + col] = f2bf(o[df][r] * inv);
        }
    }
}

// ---------------------------------------------------------------------------
// Launch
// ---------------------------------------------------------------------------
extern "C" void kernel_launch(void* const* d_in, const int* in_sizes, int n_in,
                              void* d_out, int out_size, void* d_ws, size_t ws_size,
                              hipStream_t stream) {
    (void)in_sizes; (void)n_in; (void)out_size; (void)ws_size;
    const float* in_qv  = (const float*)d_in[0];   // [2,2048,1024]
    const float* in_k   = (const float*)d_in[1];   // [2,2048,1024]
    const float* W_qv   = (const float*)d_in[2];   // [1024,2048]
    const float* W_k    = (const float*)d_in[3];   // [1024,1024]
    const float* W_proj = (const float*)d_in[4];   // [1024,1024]
    const float* b_proj = (const float*)d_in[5];   // [1024]
    float* out = (float*)d_out;

    const int B = 2, N = 2048, C = 1024;
    const int M = B * N;                           // 4096

    unsigned short* ws = (unsigned short*)d_ws;
    size_t off = 0;
    auto alloc = [&](size_t n) { unsigned short* p = ws + off; off += n; return p; };
    unsigned short* qv_b   = alloc((size_t)M * C);        // bf16 input_qv
    unsigned short* k_b    = alloc((size_t)M * C);        // bf16 input_k
    unsigned short* WqvT_b = alloc((size_t)C * 2 * C);    // bf16 [2C][C]
    unsigned short* WkT_b  = alloc((size_t)C * C);        // bf16 [C][C]
    unsigned short* WprT_b = alloc((size_t)C * C);        // bf16 [C][C]
    unsigned short* qvbuf  = alloc((size_t)M * 2 * C);    // [4096, 2048]
    unsigned short* kbuf   = alloc((size_t)M * C);        // [4096, 1024]
    unsigned short* xbuf   = alloc((size_t)M * C);        // [4096, 1024]

    // fp32 -> bf16 (activations) and fp32 -> bf16 transposed (weights)
    cvt_f32_bf16<<<(M * C) / 1024, 256, 0, stream>>>(in_qv, qv_b, M * C);
    cvt_f32_bf16<<<(M * C) / 1024, 256, 0, stream>>>(in_k, k_b, M * C);
    cvtT_f32_bf16<<<dim3((2 * C) / 64, C / 64), 256, 0, stream>>>(W_qv, WqvT_b, C, 2 * C);
    cvtT_f32_bf16<<<dim3(C / 64, C / 64), 256, 0, stream>>>(W_k, WkT_b, C, C);
    cvtT_f32_bf16<<<dim3(C / 64, C / 64), 256, 0, stream>>>(W_proj, WprT_b, C, C);

    // projections
    gemm_bf16<0><<<dim3((2 * C) / 128, M / 128), 256, 0, stream>>>(
        qv_b, WqvT_b, qvbuf, nullptr, M, 2 * C, C);
    gemm_bf16<0><<<dim3(C / 128, M / 128), 256, 0, stream>>>(
        k_b, WkT_b, kbuf, nullptr, M, C, C);

    // flash attention: grid (N/128, B*H)
    flash_attn<<<dim3(N / 128, B * 16), 256, 0, stream>>>(qvbuf, kbuf, xbuf, N);

    // output projection + bias (fp32 out)
    gemm_bf16<1><<<dim3(C / 128, M / 128), 256, 0, stream>>>(
        xbuf, WprT_b, out, b_proj, M, C, C);
}
